// MSAColumnGlobalAttention_9268539425320
// MI455X (gfx1250) — compile-verified, matrix operand order checked by reading, not run on hardware
//
#include <hip/hip_runtime.h>
#include <hip/hip_bf16.h>

typedef __attribute__((ext_vector_type(16))) __bf16 v16bf;
typedef __attribute__((ext_vector_type(8)))  float  v8f;
typedef unsigned int u32;

#define R_  384
#define S_  2048
#define C_  64
#define EPS_ 1e-5f

// pack two floats -> packed bf16x2 (RNE) ; lowers to v_cvt_pk_bf16_f32
__device__ __forceinline__ u32 pkbf(float a, float b) {
    union { __bf16 h[2]; u32 u; } x;
    x.h[0] = (__bf16)a; x.h[1] = (__bf16)b;
    return x.u;
}

// ---------------------------------------------------------------------------
// bf16 16x16x32 fragment load from a row-major bf16 [row][k] LDS buffer.
// lane -> row = lane&15 (M for A, N for B), khalf = lane>>4.
// element e maps to K = kc*32 + (e>>3)*16 + khalf*8 + (e&7)
//  => two contiguous 16B chunks: k0 = kc*32 + khalf*8 and k0+16.
// ---------------------------------------------------------------------------
__device__ __forceinline__ v16bf load_frag(const __bf16* base, int ld,
                                           int row, int kc, int khalf) {
    const uint4* p0 = (const uint4*)(base + row * ld + kc * 32 + khalf * 8);
    const uint4* p1 = (const uint4*)(base + row * ld + kc * 32 + khalf * 8 + 16);
    union { uint4 q[2]; v16bf v; } u;
    u.q[0] = *p0;
    u.q[1] = *p1;
    return u.v;
}

// ---------------------------------------------------------------------------
// Kernel 1: LayerNorm + K/V projection (WMMA) + partial sums for q_avg.
// grid = (S/64, R), block = 128 (4 waves, each wave owns a 16-token s-tile)
// ---------------------------------------------------------------------------
__global__ __launch_bounds__(128)
void msa_k1(const float* __restrict__ x, const float* __restrict__ lnw,
            const float* __restrict__ lnb, const float* __restrict__ Wk,
            const float* __restrict__ Wv, float* __restrict__ kv,
            float* __restrict__ qpart) {
    __shared__ alignas(16) __bf16 sWkv[16 * 72];   // transposed [n][k], n<8:Wk n>=8:Wv
    __shared__ float sLn[128];                     // [0:64) ln_w, [64:128) ln_b
    __shared__ alignas(16) __bf16 sxn[4][16 * 72]; // per-wave xn tile, bf16 [t][k]
    __shared__ alignas(16) float  sko[4][16 * 20]; // kv store staging [t][n]

    const int tid  = threadIdx.x;
    const int lane = tid & 31;
    const int wave = tid >> 5;
    const int r    = blockIdx.y;

    // cooperative weight load (transpose to [n][k], bf16)
#pragma unroll
    for (int i = tid; i < 1024; i += 128) {
        int k = i >> 4, n = i & 15;
        float w = (n < 8) ? Wk[k * 8 + n] : Wv[k * 8 + (n - 8)];
        sWkv[n * 72 + k] = (__bf16)w;
    }
    if (tid < 64)  sLn[tid] = lnw[tid];
    else           sLn[tid] = lnb[tid - 64];
    __syncthreads();

    const int s0   = (blockIdx.x * 4 + wave) * 16;
    const int t    = lane >> 1;     // token within tile
    const int half = lane & 1;      // which 32 channels

    const float* xrow = x + (((size_t)(s0 + t)) * R_ + r) * C_ + half * 32;
    float4 xv[8];
#pragma unroll
    for (int i = 0; i < 8; ++i) xv[i] = ((const float4*)xrow)[i];

    float s = 0.f, sq = 0.f;
#pragma unroll
    for (int i = 0; i < 8; ++i) {
        s  += xv[i].x + xv[i].y + xv[i].z + xv[i].w;
        sq += xv[i].x * xv[i].x + xv[i].y * xv[i].y +
              xv[i].z * xv[i].z + xv[i].w * xv[i].w;
    }
    float so   = __shfl_xor(s, 1);
    float sqo  = __shfl_xor(sq, 1);
    float mean = (s + so) * (1.f / 64.f);
    float var  = (sq + sqo) * (1.f / 64.f) - mean * mean;
    float rstd = rsqrtf(var + EPS_);

    // normalize + pack to bf16, 4x ds_store_b128
    {
        uint4* dst = (uint4*)&sxn[wave][t * 72 + half * 32];
#pragma unroll
        for (int i = 0; i < 4; ++i) {
            int cb = half * 32 + i * 8;
            float4 xa = xv[2 * i], xb = xv[2 * i + 1];
            float n0 = (xa.x - mean) * rstd * sLn[cb + 0] + sLn[64 + cb + 0];
            float n1 = (xa.y - mean) * rstd * sLn[cb + 1] + sLn[64 + cb + 1];
            float n2 = (xa.z - mean) * rstd * sLn[cb + 2] + sLn[64 + cb + 2];
            float n3 = (xa.w - mean) * rstd * sLn[cb + 3] + sLn[64 + cb + 3];
            float n4 = (xb.x - mean) * rstd * sLn[cb + 4] + sLn[64 + cb + 4];
            float n5 = (xb.y - mean) * rstd * sLn[cb + 5] + sLn[64 + cb + 5];
            float n6 = (xb.z - mean) * rstd * sLn[cb + 6] + sLn[64 + cb + 6];
            float n7 = (xb.w - mean) * rstd * sLn[cb + 7] + sLn[64 + cb + 7];
            uint4 w;
            w.x = pkbf(n0, n1); w.y = pkbf(n2, n3);
            w.z = pkbf(n4, n5); w.w = pkbf(n6, n7);
            dst[i] = w;
        }
    }

    // partial q sum (deterministic, per-tile): 2 channels per lane, f32 accum
    {
        float a0 = 0.f, a1 = 0.f;
#pragma unroll
        for (int tt = 0; tt < 16; ++tt) {
            union { u32 u; __bf16 h[2]; } pw;
            pw.u = *(const u32*)&sxn[wave][tt * 72 + lane * 2];
            a0 += (float)pw.h[0];
            a1 += (float)pw.h[1];
        }
        int tile = blockIdx.x * 4 + wave;
        size_t qo = (((size_t)r) * 128 + tile) * 64;
        qpart[qo + lane * 2 + 0] = a0;
        qpart[qo + lane * 2 + 1] = a1;
    }

    // K/V projection: (16x64) @ (64x16) via two bf16 WMMAs
    const int m     = lane & 15;
    const int khalf = lane >> 4;
    v16bf a0 = load_frag(sxn[wave], 72, m, 0, khalf);
    v16bf a1 = load_frag(sxn[wave], 72, m, 1, khalf);
    v16bf b0 = load_frag(sWkv, 72, m /*== n*/, 0, khalf);
    v16bf b1 = load_frag(sWkv, 72, m, 1, khalf);
    v8f acc = {};
    acc = __builtin_amdgcn_wmma_f32_16x16x32_bf16(false, a0, false, b0, (short)0, acc, false, false);
    acc = __builtin_amdgcn_wmma_f32_16x16x32_bf16(false, a1, false, b1, (short)0, acc, false, false);

    // stage D and store kv fully coalesced (2x b128 per lane)
    const int mb = khalf * 8;
#pragma unroll
    for (int v = 0; v < 8; ++v) sko[wave][(mb + v) * 20 + m] = acc[v];
    {
        int tt = lane >> 1, hh = lane & 1;
        const float4* src = (const float4*)&sko[wave][tt * 20 + hh * 8];
        float4* gd = (float4*)&kv[(((size_t)r) * S_ + s0 + tt) * 16 + hh * 8];
        gd[0] = src[0];
        gd[1] = src[1];
    }
}

// ---------------------------------------------------------------------------
// Kernel 2: q_avg reduce -> q -> streaming softmax over S -> o[r][64].
// grid = R, block = 256 (8 waves = 8 heads)
// ---------------------------------------------------------------------------
__global__ __launch_bounds__(256)
void msa_k2(const float* __restrict__ Wq, const float* __restrict__ kv,
            const float* __restrict__ qpart, float* __restrict__ obuf) {
    __shared__ float sred[4][64];
    __shared__ float sqa[64];
    __shared__ float sqh[64];

    const int r   = blockIdx.x;
    const int tid = threadIdx.x;
    const int ch  = tid & 63;
    const int grp = tid >> 6;

    float p = 0.f;
    for (int tile = grp; tile < 128; tile += 4)
        p += qpart[(((size_t)r) * 128 + tile) * 64 + ch];
    sred[grp][ch] = p;
    __syncthreads();

    if (tid < 64)
        sqa[tid] = (sred[0][tid] + sred[1][tid] + sred[2][tid] + sred[3][tid]) *
                   (1.0f / (float)S_);
    __syncthreads();

    if (tid < 64) {
        float a = 0.f;
        for (int k = 0; k < 64; ++k) a += sqa[k] * Wq[k * 64 + tid];
        sqh[tid] = a * 0.35355339059327373f;  // c^-0.5 = 1/sqrt(8)
    }
    __syncthreads();

    const int h    = tid >> 5;  // head
    const int lane = tid & 31;
    float qh[8];
#pragma unroll
    for (int j = 0; j < 8; ++j) qh[j] = sqh[h * 8 + j];

    // pass 1: max
    float mx = -1e30f;
    for (int ss = lane; ss < S_; ss += 32) {
        const float4* kr = (const float4*)&kv[(((size_t)r) * S_ + ss) * 16];
        float4 k0 = kr[0], k1 = kr[1];
        float lg = qh[0] * k0.x + qh[1] * k0.y + qh[2] * k0.z + qh[3] * k0.w +
                   qh[4] * k1.x + qh[5] * k1.y + qh[6] * k1.z + qh[7] * k1.w;
        mx = fmaxf(mx, lg);
    }
#pragma unroll
    for (int off = 16; off >= 1; off >>= 1) mx = fmaxf(mx, __shfl_xor(mx, off));

    // pass 2: sum of exp and weighted V
    float sum = 0.f;
    float oa[8] = {0.f, 0.f, 0.f, 0.f, 0.f, 0.f, 0.f, 0.f};
    for (int ss = lane; ss < S_; ss += 32) {
        const float4* kr = (const float4*)&kv[(((size_t)r) * S_ + ss) * 16];
        float4 k0 = kr[0], k1 = kr[1], v0 = kr[2], v1 = kr[3];
        float lg = qh[0] * k0.x + qh[1] * k0.y + qh[2] * k0.z + qh[3] * k0.w +
                   qh[4] * k1.x + qh[5] * k1.y + qh[6] * k1.z + qh[7] * k1.w;
        float e = __expf(lg - mx);
        sum += e;
        oa[0] += e * v0.x; oa[1] += e * v0.y; oa[2] += e * v0.z; oa[3] += e * v0.w;
        oa[4] += e * v1.x; oa[5] += e * v1.y; oa[6] += e * v1.z; oa[7] += e * v1.w;
    }
#pragma unroll
    for (int off = 16; off >= 1; off >>= 1) {
        sum += __shfl_xor(sum, off);
#pragma unroll
        for (int j = 0; j < 8; ++j) oa[j] += __shfl_xor(oa[j], off);
    }
    if (lane == 0) {
        float inv = 1.f / sum;
#pragma unroll
        for (int j = 0; j < 8; ++j) obuf[r * 64 + h * 8 + j] = oa[j] * inv;
    }
}

// ---------------------------------------------------------------------------
// Kernel 3: recompute LN, gate = sigmoid(xn@Wg+bg), u = o*gate, out = u@Wf+bf.
// grid = (S/64, R), block = 128 (4 waves, each wave a 16-token s-tile)
// ---------------------------------------------------------------------------
__global__ __launch_bounds__(128)
void msa_k3(const float* __restrict__ x, const float* __restrict__ lnw,
            const float* __restrict__ lnb, const float* __restrict__ Wg,
            const float* __restrict__ bg, const float* __restrict__ Wf,
            const float* __restrict__ bf, const float* __restrict__ obuf,
            float* __restrict__ out) {
    __shared__ alignas(16) __bf16 sWgT[64 * 72];   // transposed [n][k] bf16
    __shared__ alignas(16) __bf16 sWfT[64 * 72];   // transposed [n][k] bf16
    __shared__ float sLn[128];
    __shared__ float sbg[64];
    __shared__ float sbf[64];
    __shared__ float so_[64];
    __shared__ alignas(16) __bf16 sxn[4][16 * 72]; // xn bf16, later reused for u
    __shared__ alignas(16) float  sout[4][16 * 68];// output staging [t][c]

    const int tid  = threadIdx.x;
    const int lane = tid & 31;
    const int wave = tid >> 5;
    const int r    = blockIdx.y;

#pragma unroll
    for (int i = tid; i < 4096; i += 128) {
        int k = i >> 6, n = i & 63;
        sWgT[n * 72 + k] = (__bf16)Wg[i];
        sWfT[n * 72 + k] = (__bf16)Wf[i];
    }
    if (tid < 64)       sLn[tid] = lnw[tid];
    else                sLn[tid] = lnb[tid - 64];
    if (tid < 64)       { sbg[tid] = bg[tid]; sbf[tid] = bf[tid]; so_[tid] = obuf[r * 64 + tid]; }
    __syncthreads();

    const int s0   = (blockIdx.x * 4 + wave) * 16;
    const int t    = lane >> 1;
    const int half = lane & 1;

    const float* xrow = x + (((size_t)(s0 + t)) * R_ + r) * C_ + half * 32;
    float4 xv[8];
#pragma unroll
    for (int i = 0; i < 8; ++i) xv[i] = ((const float4*)xrow)[i];

    float s = 0.f, sq = 0.f;
#pragma unroll
    for (int i = 0; i < 8; ++i) {
        s  += xv[i].x + xv[i].y + xv[i].z + xv[i].w;
        sq += xv[i].x * xv[i].x + xv[i].y * xv[i].y +
              xv[i].z * xv[i].z + xv[i].w * xv[i].w;
    }
    float so2  = __shfl_xor(s, 1);
    float sqo  = __shfl_xor(sq, 1);
    float mean = (s + so2) * (1.f / 64.f);
    float var  = (sq + sqo) * (1.f / 64.f) - mean * mean;
    float rstd = rsqrtf(var + EPS_);

    {
        uint4* dst = (uint4*)&sxn[wave][t * 72 + half * 32];
#pragma unroll
        for (int i = 0; i < 4; ++i) {
            int cb = half * 32 + i * 8;
            float4 xa = xv[2 * i], xb = xv[2 * i + 1];
            float n0 = (xa.x - mean) * rstd * sLn[cb + 0] + sLn[64 + cb + 0];
            float n1 = (xa.y - mean) * rstd * sLn[cb + 1] + sLn[64 + cb + 1];
            float n2 = (xa.z - mean) * rstd * sLn[cb + 2] + sLn[64 + cb + 2];
            float n3 = (xa.w - mean) * rstd * sLn[cb + 3] + sLn[64 + cb + 3];
            float n4 = (xb.x - mean) * rstd * sLn[cb + 4] + sLn[64 + cb + 4];
            float n5 = (xb.y - mean) * rstd * sLn[cb + 5] + sLn[64 + cb + 5];
            float n6 = (xb.z - mean) * rstd * sLn[cb + 6] + sLn[64 + cb + 6];
            float n7 = (xb.w - mean) * rstd * sLn[cb + 7] + sLn[64 + cb + 7];
            uint4 w;
            w.x = pkbf(n0, n1); w.y = pkbf(n2, n3);
            w.z = pkbf(n4, n5); w.w = pkbf(n6, n7);
            dst[i] = w;
        }
    }

    // xn tile fully captured in A fragments (registers)
    const int m     = lane & 15;
    const int khalf = lane >> 4;
    const int mb    = khalf * 8;
    v16bf a0 = load_frag(sxn[wave], 72, m, 0, khalf);
    v16bf a1 = load_frag(sxn[wave], 72, m, 1, khalf);

    // GEMM 1: gate_in = xn @ Wg ; u = o * sigmoid(gate_in + bg) -> back to sxn (bf16)
#pragma unroll
    for (int nt = 0; nt < 4; ++nt) {
        v16bf b0 = load_frag(sWgT + nt * 16 * 72, 72, m, 0, khalf);
        v16bf b1 = load_frag(sWgT + nt * 16 * 72, 72, m, 1, khalf);
        v8f acc = {};
        acc = __builtin_amdgcn_wmma_f32_16x16x32_bf16(false, a0, false, b0, (short)0, acc, false, false);
        acc = __builtin_amdgcn_wmma_f32_16x16x32_bf16(false, a1, false, b1, (short)0, acc, false, false);
        int ncol = nt * 16 + m;
        float ov  = so_[ncol];
        float bgv = sbg[ncol];
#pragma unroll
        for (int v = 0; v < 8; ++v) {
            float g   = acc[v] + bgv;
            float sig = 1.f / (1.f + __expf(-g));
            sxn[wave][(mb + v) * 72 + ncol] = (__bf16)(ov * sig);  // in-order LDS per wave
        }
    }

    // GEMM 2: out = u @ Wf + bf  (u now lives in sxn as bf16)
    v16bf u0 = load_frag(sxn[wave], 72, m, 0, khalf);
    v16bf u1 = load_frag(sxn[wave], 72, m, 1, khalf);
#pragma unroll
    for (int nt = 0; nt < 4; ++nt) {
        v16bf b0 = load_frag(sWfT + nt * 16 * 72, 72, m, 0, khalf);
        v16bf b1 = load_frag(sWfT + nt * 16 * 72, 72, m, 1, khalf);
        v8f acc = {};
        acc = __builtin_amdgcn_wmma_f32_16x16x32_bf16(false, u0, false, b0, (short)0, acc, false, false);
        acc = __builtin_amdgcn_wmma_f32_16x16x32_bf16(false, u1, false, b1, (short)0, acc, false, false);
        int ncol = nt * 16 + m;
        float bfv = sbf[ncol];
#pragma unroll
        for (int v = 0; v < 8; ++v)
            sout[wave][(mb + v) * 68 + ncol] = acc[v] + bfv;
    }

    // coalesced store: each lane writes 32 contiguous floats (8x b128)
    {
        int tt = lane >> 1, hh = lane & 1;
        const float4* srcp = (const float4*)&sout[wave][tt * 68 + hh * 32];
        float4* gd = (float4*)&out[(((size_t)(s0 + tt)) * R_ + r) * C_ + hh * 32];
#pragma unroll
        for (int i = 0; i < 8; ++i) gd[i] = srcp[i];
    }
}

// ---------------------------------------------------------------------------
extern "C" void kernel_launch(void* const* d_in, const int* in_sizes, int n_in,
                              void* d_out, int out_size, void* d_ws, size_t ws_size,
                              hipStream_t stream) {
    (void)in_sizes; (void)n_in; (void)out_size; (void)ws_size;

    const float* x1d = (const float*)d_in[0];
    const float* lnw = (const float*)d_in[1];
    const float* lnb = (const float*)d_in[2];
    const float* Wq  = (const float*)d_in[3];
    const float* Wk  = (const float*)d_in[4];
    const float* Wv  = (const float*)d_in[5];
    const float* Wg  = (const float*)d_in[6];
    const float* bg  = (const float*)d_in[7];
    const float* Wf  = (const float*)d_in[8];
    const float* bf  = (const float*)d_in[9];
    float* out = (float*)d_out;

    float* kv    = (float*)d_ws;                        // R*S*16 floats (k|v packed)
    float* qpart = kv + (size_t)R_ * S_ * 16;           // R*128*64 floats
    float* obuf  = qpart + (size_t)R_ * 128 * 64;       // R*64 floats

    dim3 blk1(128), grd1(S_ / 64, R_);
    msa_k1<<<grd1, blk1, 0, stream>>>(x1d, lnw, lnb, Wk, Wv, kv, qpart);
    msa_k2<<<R_, 256, 0, stream>>>(Wq, kv, qpart, obuf);
    msa_k3<<<grd1, blk1, 0, stream>>>(x1d, lnw, lnb, Wg, bg, Wf, bf, obuf, out);
}
